// SlidingGaussianWindow1d_74388833567037
// MI455X (gfx1250) — matrix operand order repaired and checked by reading, not compiled, not run on hardware
//
#include <hip/hip_runtime.h>
#include <hip/hip_bf16.h>

typedef __attribute__((ext_vector_type(16))) _Float16 v16h;
typedef __attribute__((ext_vector_type(8)))  _Float16 v8h;
typedef __attribute__((ext_vector_type(8)))  float    v8f;

#define TT   256
#define CC   64
#define NWIN 224
#define YST  264   // padded t-stride (halves): 528B = 33*16B -> 16B-aligned rows, 4-bank lane skew

__global__ __launch_bounds__(256)
void sliding_corr_wmma_kernel(const float* __restrict__ x,
                              float* __restrict__ out)
{
    __shared__ __align__(16) _Float16 Yt[CC * YST];  // Y^T: [c][t] in f16 (33.8 KB)
    __shared__ float scratch[4096];                  // phase0: gw/Afull/max; later: cov 64x64
    __shared__ float wgtLds[TT];
    __shared__ float partLds[256];
    __shared__ float muLds[CC];
    __shared__ float invdLds[CC];

    const int tid = threadIdx.x;
    const int w   = blockIdx.x;   // window 0..223
    const int b   = blockIdx.y;   // batch

    float* gwLds    = scratch;         // [0..255]
    float* AfullLds = scratch + 256;   // [0..511]  (full conv, we need n in [0,510])
    float* maxLds   = scratch + 1024;  // [0..255]

    // ---------- Phase 0: window weights (1/(2a^2) scale cancels in A/A.max()) ----------
    {
        float d = (float)(tid - 128);
        gwLds[tid] = expf(-d * d);
    }
    __syncthreads();
    #pragma unroll
    for (int p = 0; p < 2; ++p) {
        int n = tid + p * 256;
        float s = 0.0f;
        if (n < 511) {
            #pragma unroll
            for (int j = 112; j < 144; ++j) {   // box b=1 on [112,144)
                int k = n - j;
                if (k >= 0 && k < 256) s += gwLds[k];
            }
        }
        if (n < 512) AfullLds[n] = s;           // Afull[511] = 0 pad
    }
    __syncthreads();
    maxLds[tid] = fmaxf(AfullLds[tid], AfullLds[tid + 256]);
    __syncthreads();
    for (int st = 128; st > 0; st >>= 1) {
        if (tid < st) maxLds[tid] = fmaxf(maxLds[tid], maxLds[tid + st]);
        __syncthreads();
    }
    {
        float invMax = 1.0f / maxLds[0];
        // W[w,t] = A[(t + S_w) mod 256], A[r] = Afull[128+r]/max, S_w = (w+1)(w-222)/2
        int S = ((w + 1) * (w - 222)) / 2;
        int r = ((tid + S) % 256 + 256) % 256;
        wgtLds[tid] = AfullLds[128 + r] * invMax;
    }
    __syncthreads();

    // ---------- Phase 1: stage Y^T = (x .* w)^T into LDS as f16 ----------
    const float4* xv = (const float4*)(x + (size_t)b * (TT * CC));
    #pragma unroll
    for (int it = 0; it < 16; ++it) {
        int f = tid + it * 256;            // float4 index 0..4095 over [t][c]
        float4 v = xv[f];
        int t  = f >> 4;
        int c0 = (f & 15) << 2;
        float wt = wgtLds[t];
        Yt[(c0 + 0) * YST + t] = (_Float16)(v.x * wt);
        Yt[(c0 + 1) * YST + t] = (_Float16)(v.y * wt);
        Yt[(c0 + 2) * YST + t] = (_Float16)(v.z * wt);
        Yt[(c0 + 3) * YST + t] = (_Float16)(v.w * wt);
    }
    __syncthreads();

    // ---------- Phase 2: per-channel means over t (from the same f16 values) ----------
    {
        int c = tid & 63;
        int q = tid >> 6;
        float s = 0.0f;
        #pragma unroll
        for (int u = 0; u < 8; ++u) {
            v8h h = *(const v8h*)(&Yt[c * YST + q * 64 + u * 8]);
            #pragma unroll
            for (int e = 0; e < 8; ++e) s += (float)h[e];
        }
        partLds[tid] = s;                  // tid == q*64 + c
    }
    __syncthreads();
    if (tid < 64) {
        muLds[tid] = (partLds[tid] + partLds[tid + 64] +
                      partLds[tid + 128] + partLds[tid + 192]) * (1.0f / 256.0f);
    }
    __syncthreads();

    // ---------- Phase 3: G = Y^T Y via v_wmma_f32_16x16x32_f16 ----------
    const int wave = tid >> 5;
    const int lane = tid & 31;
    const int nd   = lane & 15;            // row (A) / col (B,D) selector
    const int g    = lane >> 4;            // lane half-group
    const int ti   = wave >> 1;            // row tile 0..3
    const int tj0  = (wave & 1) * 2;       // col tiles tj0, tj0+1

    v8f acc0 = {}; v8f acc1 = {};
    const int arow  = (ti * 16 + nd) * YST;
    const int brow0 = ( tj0      * 16 + nd) * YST;
    const int brow1 = ((tj0 + 1) * 16 + nd) * YST;

    #pragma unroll
    for (int s = 0; s < 8; ++s) {
        const int t0 = s * 32;
        // A operand (16x32, 16-bit): lane g holds K = 16g..16g+15 contiguous
        v8h alo  = *(const v8h*)(&Yt[arow  + t0 + 16 * g]);
        v8h ahi  = *(const v8h*)(&Yt[arow  + t0 + 16 * g + 8]);
        // B operand (32x16): lane g holds K = 16g..16g+15 contiguous
        v8h b0lo = *(const v8h*)(&Yt[brow0 + t0 + 16 * g]);
        v8h b0hi = *(const v8h*)(&Yt[brow0 + t0 + 16 * g + 8]);
        v8h b1lo = *(const v8h*)(&Yt[brow1 + t0 + 16 * g]);
        v8h b1hi = *(const v8h*)(&Yt[brow1 + t0 + 16 * g + 8]);
        v16h a, b0, b1;
        #pragma unroll
        for (int e = 0; e < 8; ++e) {
            a[e]  = alo[e];   a[e + 8]  = ahi[e];
            b0[e] = b0lo[e];  b0[e + 8] = b0hi[e];
            b1[e] = b1lo[e];  b1[e + 8] = b1hi[e];
        }
        acc0 = __builtin_amdgcn_wmma_f32_16x16x32_f16(false, a, false, b0,
                                                      (short)0, acc0, false, false);
        acc1 = __builtin_amdgcn_wmma_f32_16x16x32_f16(false, a, false, b1,
                                                      (short)0, acc1, false, false);
    }

    // cov = (G - T*mu*mu^T) / (T-1); write into scratch (phase0 data is dead)
    float* covLds = scratch;
    #pragma unroll
    for (int r = 0; r < 8; ++r) {
        int ci  = ti * 16 + r + 8 * g;     // D row = r + 8*(lane>>4)
        int cj0 = tj0 * 16 + nd;           // D col = lane&15
        int cj1 = cj0 + 16;
        float mui = muLds[ci];
        covLds[ci * 64 + cj0] = (acc0[r] - 256.0f * mui * muLds[cj0]) * (1.0f / 255.0f);
        covLds[ci * 64 + cj1] = (acc1[r] - 256.0f * mui * muLds[cj1]) * (1.0f / 255.0f);
    }
    __syncthreads();

    if (tid < 64) invdLds[tid] = rsqrtf(covLds[tid * 65]);
    __syncthreads();

    // ---------- Phase 4: normalize, clamp, coalesced store ----------
    float4* outv = (float4*)(out + ((size_t)(b * NWIN + w)) * 4096);
    #pragma unroll
    for (int k2 = 0; k2 < 4; ++k2) {
        int f  = tid + k2 * 256;           // 0..1023 float4 over [ci][cj]
        int ci = f >> 4;
        int cj = (f & 15) << 2;
        float di = invdLds[ci];
        const float* row = &covLds[ci * 64 + cj];
        float4 o;
        o.x = fminf(1.0f, fmaxf(-1.0f, row[0] * di * invdLds[cj + 0]));
        o.y = fminf(1.0f, fmaxf(-1.0f, row[1] * di * invdLds[cj + 1]));
        o.z = fminf(1.0f, fmaxf(-1.0f, row[2] * di * invdLds[cj + 2]));
        o.w = fminf(1.0f, fmaxf(-1.0f, row[3] * di * invdLds[cj + 3]));
        outv[f] = o;
    }
}

extern "C" void kernel_launch(void* const* d_in, const int* in_sizes, int n_in,
                              void* d_out, int out_size, void* d_ws, size_t ws_size,
                              hipStream_t stream) {
    (void)n_in; (void)d_ws; (void)ws_size; (void)out_size;
    const float* x = (const float*)d_in[0];
    // d_in[1] = win_alpha: mathematically cancels in A / A.max(); unused.
    float* out = (float*)d_out;
    const int B = in_sizes[0] / (TT * CC);   // 16
    dim3 grid(NWIN, B, 1);
    sliding_corr_wmma_kernel<<<grid, 256, 0, stream>>>(x, out);
}
